// StateEmbeddingNet_65429531787467
// MI455X (gfx1250) — compile-verified, matrix operand order checked by reading, not run on hardware
//
#include <hip/hip_runtime.h>
#include <hip/hip_bf16.h>

#define N_NODES 50000
#define N_EDGES 100000
#define N_GRAPHS 4096
#define NEMB 64
#define NTILES (N_NODES / 16)   // 3125 exact
#define SLOPE 0.01f

typedef __attribute__((ext_vector_type(16))) _Float16 v16h;
typedef __attribute__((ext_vector_type(8)))  _Float16 v8h;
typedef __attribute__((ext_vector_type(8)))  float    v8f;

// ---------------- WMMA fragment helper (wave32, 16x16x32 f16) ----------------
// Loads an A-layout (or identically-shaped B-layout) fragment from a row-major
// f16 matrix. For A: rows are M (base = tile row 0). For B of X@W.T with W
// row-major [O,K] f16: rows are N (base = Wh + n0*ld).
// Lane<16: K = {0..7, 16..23}; lane>=16: K = {8..15, 24..31}.
__device__ __forceinline__ v16h load_frag16(const _Float16* base, int ld, int k0, int lane) {
    int m  = lane & 15;
    int kh = (lane >> 4) * 8;
    const _Float16* p = base + m * ld + k0;
    v8h lo = *(const v8h*)(p + kh);
    v8h hi = *(const v8h*)(p + 16 + kh);
    v16h a;
#pragma unroll
    for (int j = 0; j < 8; ++j) { a[j] = lo[j]; a[8 + j] = hi[j]; }
    return a;
}

#define WMMA_F16(a, b, c) \
    __builtin_amdgcn_wmma_f32_16x16x32_f16(false, (a), false, (b), (short)0, (c), false, false)

__device__ __forceinline__ float leaky(float v) { return v > 0.0f ? v : SLOPE * v; }
__device__ __forceinline__ float sigm(float v)  { return 1.0f / (1.0f + __expf(-v)); }

// ---------------- kernels ----------------
__global__ void k_zero(float* p, int n) {
    int i = blockIdx.x * 256 + threadIdx.x;
    if (i < n) p[i] = 0.0f;
}

// one-shot: convert all weights to f16 (root also transposed so B frags are
// contiguous in k). 192*64 = 12288 threads cover everything.
__global__ void k_prep(const float* __restrict__ w1, const float* __restrict__ w2,
                       const float* __restrict__ root,
                       const float* __restrict__ wih, const float* __restrict__ whh,
                       _Float16* __restrict__ w1h, _Float16* __restrict__ w2h,
                       _Float16* __restrict__ rootTh,
                       _Float16* __restrict__ wihh, _Float16* __restrict__ whhh) {
    int i = blockIdx.x * 256 + threadIdx.x;
    if (i < 64 * 128) w1h[i] = (_Float16)w1[i];
    if (i < 64 * 64) {
        w2h[i] = (_Float16)w2[i];
        int r = i >> 6, c = i & 63;
        rootTh[c * 64 + r] = (_Float16)root[i];   // rootT[n][k] = root[k][n]
    }
    if (i < 192 * 64) {
        wihh[i] = (_Float16)wih[i];
        whhh[i] = (_Float16)whh[i];
    }
}

// initial embedding: h = leaky(cat(blockemb[x], vec[batch]) @ w1.T + b1) @ w2.T + b2
__global__ void __launch_bounds__(128)
k_init(const int* __restrict__ x, const int* __restrict__ batch,
       const float* __restrict__ vec, const float* __restrict__ blockemb,
       const _Float16* __restrict__ w1h, const float* __restrict__ b1,
       const _Float16* __restrict__ w2h, const float* __restrict__ b2,
       float* __restrict__ h32, _Float16* __restrict__ h16) {
    __shared__ _Float16 lds[4][16 * NEMB];
    int wave = threadIdx.x >> 5, lane = threadIdx.x & 31;
    int tile = blockIdx.x * 4 + wave;
    if (tile >= NTILES) return;            // wave-uniform guard
    int row0 = tile * 16;
    int m = lane & 15, kh = (lane >> 4) * 8, mb = (lane >> 4) * 8, n = lane & 15;
    int node = row0 + m;
    const float* pb = blockemb + x[node] * NEMB;
    const float* pv = vec + batch[node] * NEMB;

    // A fragments of the 128-wide concat input: 4 chunks of K=32 (f32 gather + cvt,
    // runs once per node per call)
    v16h A[4];
#pragma unroll
    for (int c = 0; c < 4; ++c) {
        int k0 = c * 32;
        const float* s = (c < 2) ? (pb + k0) : (pv + (k0 - 64));
        v16h a;
#pragma unroll
        for (int j = 0; j < 8; ++j) {
            a[j]     = (_Float16)s[kh + j];
            a[8 + j] = (_Float16)s[16 + kh + j];
        }
        A[c] = a;
    }
    // layer1: [16,128] @ w1.T -> [16,64], leaky, stage f16 in LDS
#pragma unroll
    for (int t = 0; t < 4; ++t) {
        v8f acc;
#pragma unroll
        for (int r = 0; r < 8; ++r) acc[r] = b1[t * 16 + n];
#pragma unroll
        for (int c = 0; c < 4; ++c)
            acc = WMMA_F16(A[c], load_frag16(w1h + (t * 16) * 128, 128, c * 32, lane), acc);
#pragma unroll
        for (int r = 0; r < 8; ++r)
            lds[wave][(mb + r) * NEMB + t * 16 + n] = (_Float16)leaky(acc[r]);
    }
    // layer2: [16,64] @ w2.T -> [16,64]
    v16h A2[2];
#pragma unroll
    for (int c = 0; c < 2; ++c) A2[c] = load_frag16(lds[wave], NEMB, c * 32, lane);
#pragma unroll
    for (int t = 0; t < 4; ++t) {
        v8f acc;
#pragma unroll
        for (int r = 0; r < 8; ++r) acc[r] = b2[t * 16 + n];
#pragma unroll
        for (int c = 0; c < 2; ++c)
            acc = WMMA_F16(A2[c], load_frag16(w2h + (t * 16) * 64, 64, c * 32, lane), acc);
#pragma unroll
        for (int r = 0; r < 8; ++r) {
            int idx = (row0 + mb + r) * NEMB + t * 16 + n;
            h32[idx] = acc[r];
            h16[idx] = (_Float16)acc[r];
        }
    }
}

__global__ void k_deg(const int* __restrict__ dst, float* __restrict__ deg) {
    int e = blockIdx.x * 256 + threadIdx.x;
    if (e < N_EDGES) atomicAdd(&deg[dst[e]], 1.0f);
}

// rank-1 edge message: msg = (h[src] . bondemb[a0]) * bondemb[a1], scatter-add at dst
__global__ void __launch_bounds__(256)
k_edge(const float* __restrict__ h32, const int* __restrict__ ei,
       const int* __restrict__ eattr, const float* __restrict__ bondemb,
       float* __restrict__ agg) {
    int wave = threadIdx.x >> 5, lane = threadIdx.x & 31;
    int e = blockIdx.x * 8 + wave;
    if (e >= N_EDGES) return;
    int s = ei[e], d = ei[N_EDGES + e];
    int a0 = eattr[2 * e], a1 = eattr[2 * e + 1];
    const float* hs = h32 + s * NEMB;
    const float* e0 = bondemb + a0 * NEMB;
    const float* e1 = bondemb + a1 * NEMB;
    float part = hs[lane] * e0[lane] + hs[lane + 32] * e0[lane + 32];
#pragma unroll
    for (int off = 16; off > 0; off >>= 1) part += __shfl_xor(part, off, 32);
    atomicAdd(&agg[d * NEMB + lane],      part * e1[lane]);
    atomicAdd(&agg[d * NEMB + lane + 32], part * e1[lane + 32]);
}

// one conv + GRU step: m = leaky(agg/deg + h@root + cb); GRU(m, h) -> h
__global__ void __launch_bounds__(128)
k_step(const float* __restrict__ agg, const float* __restrict__ deg,
       const _Float16* __restrict__ rootTh, const float* __restrict__ cb,
       const _Float16* __restrict__ wihh, const _Float16* __restrict__ whhh,
       const float* __restrict__ bih, const float* __restrict__ bhh,
       float* __restrict__ h32, _Float16* __restrict__ h16) {
    __shared__ _Float16 lds[4][16 * NEMB];
    int wave = threadIdx.x >> 5, lane = threadIdx.x & 31;
    int tile = blockIdx.x * 4 + wave;
    if (tile >= NTILES) return;
    int row0 = tile * 16;
    int mb = (lane >> 4) * 8, n = lane & 15;

    // A fragments of current h (f16, row-major)
    v16h Ah[2];
#pragma unroll
    for (int c = 0; c < 2; ++c) Ah[c] = load_frag16(h16 + row0 * NEMB, NEMB, c * 32, lane);

    // m = leaky(agg/deg + cb + h @ root), staged f16 in LDS
#pragma unroll
    for (int t = 0; t < 4; ++t) {
        v8f acc;
#pragma unroll
        for (int r = 0; r < 8; ++r) {
            int row = row0 + mb + r;
            float dg = deg[row]; dg = dg < 1.0f ? 1.0f : dg;
            acc[r] = agg[row * NEMB + t * 16 + n] / dg + cb[t * 16 + n];
        }
#pragma unroll
        for (int c = 0; c < 2; ++c)
            acc = WMMA_F16(Ah[c], load_frag16(rootTh + (t * 16) * 64, 64, c * 32, lane), acc);
#pragma unroll
        for (int r = 0; r < 8; ++r)
            lds[wave][(mb + r) * NEMB + t * 16 + n] = (_Float16)leaky(acc[r]);
    }
    v16h Am[2];
#pragma unroll
    for (int c = 0; c < 2; ++c) Am[c] = load_frag16(lds[wave], NEMB, c * 32, lane);

    // GRU gates, one 16-column slab at a time (r/z/n gate rows of w_ih/w_hh)
#pragma unroll
    for (int t = 0; t < 4; ++t) {
        v8f gir, giz, gin, ghr, ghz, ghn;
#pragma unroll
        for (int r = 0; r < 8; ++r) {
            int col = t * 16 + n;
            gir[r] = bih[col]; giz[r] = bih[64 + col]; gin[r] = bih[128 + col];
            ghr[r] = bhh[col]; ghz[r] = bhh[64 + col]; ghn[r] = bhh[128 + col];
        }
#pragma unroll
        for (int c = 0; c < 2; ++c) {
            int k0 = c * 32;
            gir = WMMA_F16(Am[c], load_frag16(wihh + (t * 16) * 64,         64, k0, lane), gir);
            giz = WMMA_F16(Am[c], load_frag16(wihh + (64 + t * 16) * 64,    64, k0, lane), giz);
            gin = WMMA_F16(Am[c], load_frag16(wihh + (128 + t * 16) * 64,   64, k0, lane), gin);
            ghr = WMMA_F16(Ah[c], load_frag16(whhh + (t * 16) * 64,         64, k0, lane), ghr);
            ghz = WMMA_F16(Ah[c], load_frag16(whhh + (64 + t * 16) * 64,    64, k0, lane), ghz);
            ghn = WMMA_F16(Ah[c], load_frag16(whhh + (128 + t * 16) * 64,   64, k0, lane), ghn);
        }
#pragma unroll
        for (int r = 0; r < 8; ++r) {
            int idx = (row0 + mb + r) * NEMB + t * 16 + n;
            float hold = h32[idx];
            float rr = sigm(gir[r] + ghr[r]);
            float zz = sigm(giz[r] + ghz[r]);
            float nn = tanhf(gin[r] + rr * ghn[r]);
            float hn = (1.0f - zz) * nn + zz * hold;
            h32[idx] = hn;
            h16[idx] = (_Float16)hn;
        }
    }
}

__global__ void __launch_bounds__(256)
k_pool(const float* __restrict__ h32, const int* __restrict__ batch,
       float* __restrict__ gsum, float* __restrict__ gcnt) {
    int wave = threadIdx.x >> 5, lane = threadIdx.x & 31;
    int i = blockIdx.x * 8 + wave;
    if (i >= N_NODES) return;
    int g = batch[i];
    atomicAdd(&gsum[g * NEMB + lane],      h32[i * NEMB + lane]);
    atomicAdd(&gsum[g * NEMB + lane + 32], h32[i * NEMB + lane + 32]);
    if (lane == 0) atomicAdd(&gcnt[g], 1.0f);
}

__global__ void k_div(const float* __restrict__ gsum, const float* __restrict__ gcnt,
                      float* __restrict__ out) {
    int i = blockIdx.x * 256 + threadIdx.x;
    if (i < N_GRAPHS * NEMB) {
        float c = gcnt[i >> 6];
        out[i] = gsum[i] / (c < 1.0f ? 1.0f : c);
    }
}

// ---------------- host side ----------------
extern "C" void kernel_launch(void* const* d_in, const int* in_sizes, int n_in,
                              void* d_out, int out_size, void* d_ws, size_t ws_size,
                              hipStream_t stream) {
    const int*   x        = (const int*)d_in[0];
    const int*   ei       = (const int*)d_in[1];    // [2, E]: src then dst
    const int*   eattr    = (const int*)d_in[2];    // [E, 2]
    const int*   batch    = (const int*)d_in[3];
    const float* vec      = (const float*)d_in[4];
    const float* blockemb = (const float*)d_in[5];
    const float* bondemb  = (const float*)d_in[6];
    const float* w1       = (const float*)d_in[7];
    const float* b1       = (const float*)d_in[8];
    const float* w2       = (const float*)d_in[9];
    const float* b2       = (const float*)d_in[10];
    const float* root     = (const float*)d_in[11];
    const float* cb       = (const float*)d_in[12];
    const float* wih      = (const float*)d_in[13];
    const float* whh      = (const float*)d_in[14];
    const float* bih      = (const float*)d_in[15];
    const float* bhh      = (const float*)d_in[16];
    float* out = (float*)d_out;

    float* ws   = (float*)d_ws;
    float* h32  = ws;                                // N*64 f32
    float* agg  = h32 + (size_t)N_NODES * NEMB;      // N*64 f32
    float* deg  = agg + (size_t)N_NODES * NEMB;      // N   f32
    float* gsum = deg + N_NODES;                     // G*64 f32
    float* gcnt = gsum + (size_t)N_GRAPHS * NEMB;    // G   f32
    _Float16* h16    = (_Float16*)(gcnt + N_GRAPHS); // N*64 f16 (16B aligned)
    _Float16* w1h    = h16 + (size_t)N_NODES * NEMB; // 64*128
    _Float16* w2h    = w1h + 64 * 128;               // 64*64
    _Float16* rootTh = w2h + 64 * 64;                // 64*64 (transposed)
    _Float16* wihh   = rootTh + 64 * 64;             // 192*64
    _Float16* whhh   = wihh + 192 * 64;              // 192*64

    const int tileBlocks = (NTILES + 3) / 4;         // 782

    k_prep<<<(192 * 64 + 255) / 256, 256, 0, stream>>>(w1, w2, root, wih, whh,
                                                       w1h, w2h, rootTh, wihh, whhh);
    k_zero<<<(N_NODES + 255) / 256, 256, 0, stream>>>(deg, N_NODES);
    k_init<<<tileBlocks, 128, 0, stream>>>(x, batch, vec, blockemb, w1h, b1, w2h, b2, h32, h16);
    k_deg<<<(N_EDGES + 255) / 256, 256, 0, stream>>>(ei + N_EDGES, deg);

    for (int s = 0; s < 4; ++s) {
        k_zero<<<(N_NODES * NEMB + 255) / 256, 256, 0, stream>>>(agg, N_NODES * NEMB);
        k_edge<<<(N_EDGES + 7) / 8, 256, 0, stream>>>(h32, ei, eattr, bondemb, agg);
        k_step<<<tileBlocks, 128, 0, stream>>>(agg, deg, rootTh, cb, wihh, whhh, bih, bhh, h32, h16);
    }

    k_zero<<<(N_GRAPHS * NEMB + N_GRAPHS + 255) / 256, 256, 0, stream>>>(gsum, N_GRAPHS * NEMB + N_GRAPHS);
    k_pool<<<(N_NODES + 7) / 8, 256, 0, stream>>>(h32, batch, gsum, gcnt);
    k_div<<<(N_GRAPHS * NEMB + 255) / 256, 256, 0, stream>>>(gsum, gcnt, out);
}
